// AttentionBlock_4088808866059
// MI455X (gfx1250) — matrix operand order repaired and checked by reading, not compiled, not run on hardware
//
#include <hip/hip_runtime.h>
#include <hip/hip_bf16.h>

typedef __attribute__((ext_vector_type(16))) _Float16 v16h;
typedef __attribute__((ext_vector_type(8)))  _Float16 v8h;
typedef __attribute__((ext_vector_type(2)))  _Float16 h2;
typedef __attribute__((ext_vector_type(8)))  float    v8f;
typedef __attribute__((ext_vector_type(4)))  unsigned int u32x4;
typedef __attribute__((ext_vector_type(8)))  int          i32x8;
typedef __attribute__((ext_vector_type(4)))  int          i32x4;

#define BATCH 2
#define CCH   128
#define SSP   4096
#define NH    4
#define DH    32
#define NG    32

// Tensor Data Mover path: this toolchain exposes the 6-arg builtin
// (u32x4 g0, i32x8 g1, i32x4 g2, i32x4 g3, i32x8 extra, i32 cpol).
#if defined(__has_builtin)
#  if __has_builtin(__builtin_amdgcn_tensor_load_to_lds) && \
      __has_builtin(__builtin_amdgcn_s_wait_tensorcnt)
#    define USE_TDM 1
#  endif
#endif
#ifndef USE_TDM
#  define USE_TDM 0
#endif

union H2I { h2 h; int i; };

// ---------------------------------------------------------------- GroupNorm
// one block per (batch, group): 4 channels * 4096 = 16384 elements
__global__ void groupnorm_kernel(const float* __restrict__ x,
                                 const float* __restrict__ gw,
                                 const float* __restrict__ gb,
                                 _Float16* __restrict__ h)
{
    const int b = blockIdx.x >> 5;
    const int g = blockIdx.x & 31;
    const int t = threadIdx.x;
    const int GSZ = (CCH / NG) * SSP; // 16384
    const size_t base = ((size_t)b * CCH + (size_t)g * (CCH / NG)) * SSP;

    __shared__ float ssum[256];
    __shared__ float ssq[256];

    float sum = 0.f, sq = 0.f;
    for (int i = 0; i < GSZ / 256; ++i) {
        float v = x[base + t + i * 256];
        sum += v; sq += v * v;
    }
    ssum[t] = sum; ssq[t] = sq;
    __syncthreads();
    for (int off = 128; off > 0; off >>= 1) {
        if (t < off) { ssum[t] += ssum[t + off]; ssq[t] += ssq[t + off]; }
        __syncthreads();
    }
    const float mean = ssum[0] * (1.f / GSZ);
    const float var  = ssq[0]  * (1.f / GSZ) - mean * mean;
    const float rs   = rsqrtf(var + 1e-5f);

    for (int i = 0; i < GSZ / 256; ++i) {
        int pos = t + i * 256;
        int c = g * (CCH / NG) + (pos >> 12);
        float v = (x[base + pos] - mean) * rs * gw[c] + gb[c];
        h[base + pos] = (_Float16)v;
    }
}

// ---------------------------------------------------------------- WMMA GEMM
// Y[b][o][s] = sum_c W[o][c] * X[b][c][s] + bias[o] (+ residual, MODE 1)
// One wave computes one 16(O) x 16(S) tile; K=128 in 4 x wmma 16x16x32 f16.
template <int MODE>
__global__ void wmma_gemm_kernel(const float* __restrict__ W,
                                 const _Float16* __restrict__ X,
                                 const float* __restrict__ bias,
                                 const float* __restrict__ res,
                                 _Float16* __restrict__ out16,
                                 float* __restrict__ out32,
                                 int O)
{
    const int lane = threadIdx.x & 31;
    const int wid  = threadIdx.x >> 5;
    const int n  = lane & 15;
    const int hi = lane >> 4;

    const int tile = blockIdx.x * 4 + wid;
    const int tpb  = (O / 16) * (SSP / 16);
    const int b    = tile / tpb;
    const int r0   = tile % tpb;
    const int o0   = (r0 / (SSP / 16)) * 16;
    const int s0   = (r0 % (SSP / 16)) * 16;

    const _Float16* Xb = X + (size_t)b * CCH * SSP;
    const float*    Wr = W + (size_t)(o0 + n) * CCH;

    v8f acc = {};
    for (int kc = 0; kc < CCH; kc += 32) {
        // prefetch next B panel (speculative; dropped if OOB)
        __builtin_prefetch(&Xb[(size_t)(kc + 32 + hi * 16) * SSP + s0 + n], 0, 0);
        v16h a, bb;
#pragma unroll
        for (int j = 0; j < 8; ++j) a[j]     = (_Float16)Wr[kc + hi * 8 + j];
#pragma unroll
        for (int j = 0; j < 8; ++j) a[8 + j] = (_Float16)Wr[kc + 16 + hi * 8 + j];
#pragma unroll
        for (int j = 0; j < 16; ++j)
            bb[j] = Xb[(size_t)(kc + (j < 8 ? hi * 8 + j : 16 + hi * 8 + (j - 8))) * SSP + s0 + n];
        acc = __builtin_amdgcn_wmma_f32_16x16x32_f16(false, a, false, bb,
                                                     (short)0, acc, false, false);
    }

#pragma unroll
    for (int r = 0; r < 8; ++r) {
        const int oc = o0 + r + 8 * hi;
        const size_t idx = ((size_t)b * O + oc) * SSP + s0 + n;
        float v = acc[r] + bias[oc];
        if (MODE == 0) out16[idx] = (_Float16)v;
        else           out32[idx] = v + res[idx];
    }
}

// ---------------------------------------------------------------- Flash attention
// qkv layout: [b][3*128][s], channel = qkv_idx*128 + head*32 + d.
// Block = 4 waves; each wave owns 16 query rows; block streams 32-key K/V
// chunks through LDS. V tile is DMA'd by the Tensor Data Mover (TDM) with
// hardware row padding to the 40-half pitch; K is staged cooperatively.
__global__ void flash_attn_kernel(const _Float16* __restrict__ qkv,
                                  _Float16* __restrict__ aout)
{
    __shared__ _Float16 kbuf[32][40];     // [key][d]   (padded rows, 16B-aligned)
    __shared__ _Float16 vbufT[32][40];    // [d][key]   (TDM pads rows to 40)
    __shared__ _Float16 pbuf[4][16][40];  // per-wave P staging: [m][key]

    const int tid  = threadIdx.x;
    const int lane = tid & 31;
    const int w    = tid >> 5;
    const int n  = lane & 15;
    const int hi = lane >> 4;

    const int bh   = blockIdx.x >> 6;      // 64 query blocks per (b,h)
    const int qblk = blockIdx.x & 63;
    const int b = bh >> 2;
    const int h = bh & 3;
    const int q0 = qblk * 64 + w * 16;

    const _Float16* qbase = qkv + ((size_t)b * 3 * CCH + 0 * CCH + h * DH) * SSP;
    const _Float16* kbase = qkv + ((size_t)b * 3 * CCH + 1 * CCH + h * DH) * SSP;
    const _Float16* vbase = qkv + ((size_t)b * 3 * CCH + 2 * CCH + h * DH) * SSP;

    // Q tile (16 queries x 32 dims) in A layout, loaded once
    v16h aq;
#pragma unroll
    for (int j = 0; j < 8; ++j) aq[j]     = qbase[(size_t)(hi * 8 + j) * SSP + q0 + n];
#pragma unroll
    for (int j = 0; j < 8; ++j) aq[8 + j] = qbase[(size_t)(16 + hi * 8 + j) * SSP + q0 + n];

    // all-ones B operand: D = P x ones accumulates row sums of P
    v16h bones;
#pragma unroll
    for (int j = 0; j < 16; ++j) bones[j] = (_Float16)1.0f;

    float mrow[8];
    v8f o0 = {}, o1 = {}, lacc = {};
#pragma unroll
    for (int r = 0; r < 8; ++r) mrow[r] = -1e30f;

    // (1/sqrt(32)) * log2(e): softmax in base 2 -> raw v_exp_f32
    const float scale2 = 0.17677669529663687f * 1.4426950408889634f;

#if USE_TDM
    // D# group1 is chunk-invariant: data_size=2B, pad after every 16 DWORDs
    // (one 32-half row) by 4 DWORDs -> LDS pitch 40 halves; 32x32 tile,
    // tensor dim0 stride = 4096 elements.
    i32x8 g1;
    g1[0] = (int)((1u << 16)      // data_size = 2 bytes
                | (1u << 20)      // pad_enable
                | (3u << 22)      // pad_interval: 16 DWORDs
                | (3u << 25));    // pad_amount: 4 DWORDs
    g1[1] = (int)(32u << 16);     // tensor_dim0 = 32
    g1[2] = (int)(32u << 16);     // tensor_dim1 = 32
    g1[3] = (int)(32u << 16);     // tile_dim0 = 32
    g1[4] = 32;                   // tile_dim1 = 32 (tile_dim2 = 0)
    g1[5] = (int)SSP;             // tensor_dim0_stride = 4096
    g1[6] = 0;
    g1[7] = 0;
    const i32x4 gz4 = {0, 0, 0, 0};
    const i32x8 gz8 = {0, 0, 0, 0, 0, 0, 0, 0};
    const unsigned int vlds = (unsigned int)(uintptr_t)&vbufT[0][0];
#endif

    for (int kc = 0; kc < SSP; kc += 32) {
        __syncthreads();
#if USE_TDM
        if (w == 0) {   // one wave issues the async V-tile DMA (EXEC ignored)
            unsigned long long ga = (unsigned long long)(uintptr_t)(vbase + kc);
            u32x4 g0;
            g0[0] = 1u;                               // count = 1 (valid)
            g0[1] = vlds;                             // lds_addr
            g0[2] = (unsigned int)ga;                 // global_addr[31:0]
            g0[3] = (unsigned int)((ga >> 32) & 0x01FFFFFFu)
                  | 0x80000000u;                      // type = 2 (image)
            __builtin_amdgcn_tensor_load_to_lds(g0, g1, gz4, gz4, gz8, 0);
        }
#endif
        {   // cooperative K staging: 128 threads x 8 keys (scatter to [key][d])
            const int d  = tid >> 2;
            const int kq = tid & 3;
            const _Float16* kp = kbase + (size_t)d * SSP + kc + kq * 8;
            __builtin_prefetch(kp + 32, 0, 0);   // next chunk (speculative)
#if !USE_TDM
            const _Float16* vp = vbase + (size_t)d * SSP + kc + kq * 8;
            __builtin_prefetch(vp + 32, 0, 0);
#endif
#pragma unroll
            for (int kk = 0; kk < 8; ++kk) {
                kbuf[kq * 8 + kk][d] = kp[kk];
#if !USE_TDM
                vbufT[d][kq * 8 + kk] = vp[kk];
#endif
            }
        }
#if USE_TDM
        if (w == 0) __builtin_amdgcn_s_wait_tensorcnt(0);
#endif
        __syncthreads();

        // ---- scores: 2 WMMAs for 16 queries x 32 keys
        v16h bk0, bk1;
#pragma unroll
        for (int j = 0; j < 16; ++j) {
            bk0[j] = kbuf[n][(j < 8 ? hi * 8 + j : 16 + hi * 8 + (j - 8))];
            bk1[j] = kbuf[16 + n][(j < 8 ? hi * 8 + j : 16 + hi * 8 + (j - 8))];
        }
        v8f s0v = {}, s1v = {};
        s0v = __builtin_amdgcn_wmma_f32_16x16x32_f16(false, aq, false, bk0,
                                                     (short)0, s0v, false, false);
        s1v = __builtin_amdgcn_wmma_f32_16x16x32_f16(false, aq, false, bk1,
                                                     (short)0, s1v, false, false);

        // ---- row max: packed f16 butterfly over the 16-lane groups
        float v0p[8], v1p[8];
        H2I pk[4];
#pragma unroll
        for (int r = 0; r < 8; ++r) {
            v0p[r] = s0v[r] * scale2;
            v1p[r] = s1v[r] * scale2;
        }
#pragma unroll
        for (int r = 0; r < 4; ++r) {
            h2 p;
            p[0] = (_Float16)fmaxf(v0p[r],     v1p[r]);
            p[1] = (_Float16)fmaxf(v0p[r + 4], v1p[r + 4]);
            pk[r].h = p;
        }
#pragma unroll
        for (int msk = 1; msk <= 8; msk <<= 1) {
#pragma unroll
            for (int r = 0; r < 4; ++r) {
                H2I o; o.i = __shfl_xor(pk[r].i, msk);
                pk[r].h = __builtin_elementwise_max(pk[r].h, o.h);
            }
        }

        // ---- rescale accumulators, exponentiate, stage P
#pragma unroll
        for (int r = 0; r < 8; ++r) {
            float mx = (float)((r < 4) ? pk[r].h[0] : pk[r - 4].h[1]);
            float mnew = fmaxf(mrow[r], mx);
            float corr = exp2f(mrow[r] - mnew);
            float p0 = exp2f(v0p[r] - mnew);
            float p1 = exp2f(v1p[r] - mnew);
            mrow[r] = mnew;
            o0[r]   *= corr;
            o1[r]   *= corr;
            lacc[r] *= corr;
            pbuf[w][r + 8 * hi][n]      = (_Float16)p0;
            pbuf[w][r + 8 * hi][16 + n] = (_Float16)p1;
        }

        // ---- P * V (+ row sums): restage P (D-layout -> A-layout) via LDS
        v16h ap, bv0, bv1;
#pragma unroll
        for (int j = 0; j < 8; ++j) {
            ap[j]     = pbuf[w][n][hi * 8 + j];
            ap[8 + j] = pbuf[w][n][16 + hi * 8 + j];
        }
#pragma unroll
        for (int j = 0; j < 16; ++j) {
            bv0[j] = vbufT[n][(j < 8 ? hi * 8 + j : 16 + hi * 8 + (j - 8))];
            bv1[j] = vbufT[16 + n][(j < 8 ? hi * 8 + j : 16 + hi * 8 + (j - 8))];
        }
        o0   = __builtin_amdgcn_wmma_f32_16x16x32_f16(false, ap, false, bv0,
                                                      (short)0, o0, false, false);
        o1   = __builtin_amdgcn_wmma_f32_16x16x32_f16(false, ap, false, bv1,
                                                      (short)0, o1, false, false);
        lacc = __builtin_amdgcn_wmma_f32_16x16x32_f16(false, ap, false, bones,
                                                      (short)0, lacc, false, false);
    }

    _Float16* ob = aout + (size_t)b * CCH * SSP;
#pragma unroll
    for (int r = 0; r < 8; ++r) {
        float inv = 1.f / lacc[r];
        int sidx = q0 + r + 8 * hi;
        ob[(size_t)(h * DH + n) * SSP + sidx]      = (_Float16)(o0[r] * inv);
        ob[(size_t)(h * DH + 16 + n) * SSP + sidx] = (_Float16)(o1[r] * inv);
    }
}

// ---------------------------------------------------------------- launch
extern "C" void kernel_launch(void* const* d_in, const int* in_sizes, int n_in,
                              void* d_out, int out_size, void* d_ws, size_t ws_size,
                              hipStream_t stream)
{
    const float* x      = (const float*)d_in[0];
    const float* norm_w = (const float*)d_in[1];
    const float* norm_b = (const float*)d_in[2];
    const float* qkv_w  = (const float*)d_in[3];
    const float* qkv_b  = (const float*)d_in[4];
    const float* proj_w = (const float*)d_in[5];
    const float* proj_b = (const float*)d_in[6];
    float* out = (float*)d_out;

    char* ws = (char*)d_ws;
    _Float16* h16   = (_Float16*)(ws);                              // 2 MB
    _Float16* qkv16 = (_Float16*)(ws + (size_t)2 * 1024 * 1024);    // 6 MB
    _Float16* at16  = (_Float16*)(ws + (size_t)8 * 1024 * 1024);    // 2 MB

    // 1) GroupNorm -> f16 activations
    groupnorm_kernel<<<BATCH * NG, 256, 0, stream>>>(x, norm_w, norm_b, h16);

    // 2) QKV projection: tiles = 2 * (384/16) * (4096/16) = 12288, 4 waves/block
    {
        int blocks = (BATCH * (3 * CCH / 16) * (SSP / 16)) / 4; // 3072
        wmma_gemm_kernel<0><<<blocks, 128, 0, stream>>>(
            qkv_w, h16, qkv_b, nullptr, qkv16, nullptr, 3 * CCH);
    }

    // 3) Flash attention: one block (4 waves) per 64 query rows per (b,h)
    flash_attn_kernel<<<BATCH * NH * (SSP / 64), 128, 0, stream>>>(qkv16, at16);

    // 4) Output projection + bias + residual -> f32 out
    {
        int blocks = (BATCH * (CCH / 16) * (SSP / 16)) / 4; // 1024
        wmma_gemm_kernel<1><<<blocks, 128, 0, stream>>>(
            proj_w, at16, proj_b, x, nullptr, out, CCH);
    }
}
